// YatCausalAttention_10187662426973
// MI455X (gfx1250) — matrix operand order repaired and checked by reading, not compiled, not run on hardware
//
#include <hip/hip_runtime.h>

typedef __attribute__((ext_vector_type(16))) __bf16 v16bf;
typedef __attribute__((ext_vector_type(8)))  __bf16 v8bf;
typedef __attribute__((ext_vector_type(8)))  float  v8f;

union BF16x16 { v16bf v; v8bf h[2]; };

#define EPS_  1e-6f
#define B_    2
#define T_    2048
#define C_    1024
#define H_    16
#define D_    64

// ---------------------------------------------------------------------------
// WMMA fragment loaders (wave32, bf16 16x16x32, A/B row-major "A·B^T" form)
// A frag 16x32: lane&15 = row M; K halves at (lane>>4)*8 and +16.
// B frag 32x16 from row-major [N][K]: lane&15 = row N; K halves at (lane>>4)*8 and +16.
// ---------------------------------------------------------------------------
__device__ __forceinline__ v16bf load_frag_a(const __bf16* base, int ld, int lane) {
  const __bf16* p = base + (size_t)(lane & 15) * ld + ((lane >> 4) << 3);
  BF16x16 r;
  r.h[0] = *(const v8bf*)(p);
  r.h[1] = *(const v8bf*)(p + 16);
  return r.v;
}

__device__ __forceinline__ v16bf load_frag_b(const __bf16* base, int ld, int lane) {
  const __bf16* p = base + (size_t)(lane & 15) * ld + ((lane >> 4) << 3);
  BF16x16 r;
  r.h[0] = *(const v8bf*)(p);
  r.h[1] = *(const v8bf*)(p + 16);
  return r.v;
}

// ---------------------------------------------------------------------------
// Prep kernels: fp32 -> bf16 convert, and transpose-convert (for W^T / V^T)
// ---------------------------------------------------------------------------
__global__ __launch_bounds__(256) void f32_to_bf16(const float* __restrict__ s,
                                                   __bf16* __restrict__ d, int n) {
  int i = blockIdx.x * blockDim.x + threadIdx.x;
  if (i < n) d[i] = (__bf16)s[i];
}

__global__ __launch_bounds__(256) void transpose_to_bf16(const float* __restrict__ s,
                                                         __bf16* __restrict__ d,
                                                         int R, int Cc) {
  int i = blockIdx.x * blockDim.x + threadIdx.x;
  if (i < R * Cc) {
    int r = i / Cc, c = i % Cc;
    d[(size_t)c * R + r] = (__bf16)s[i];
  }
}

// From qkv bf16 [B*T, 3C]: per (b,h,t) compute qn/kn = |q|^2,|k|^2 and v^T [B*H, D, T]
__global__ __launch_bounds__(256) void prep_heads(const __bf16* __restrict__ qkv,
                                                  __bf16* __restrict__ vT,
                                                  float* __restrict__ qn,
                                                  float* __restrict__ kn) {
  int idx = blockIdx.x * blockDim.x + threadIdx.x;
  if (idx >= B_ * H_ * T_) return;
  int t  = idx % T_;
  int bh = idx / T_;
  int b  = bh / H_, h = bh % H_;
  const __bf16* row = qkv + (size_t)(b * T_ + t) * (3 * C_) + h * D_;
  float sq = 0.f, sk = 0.f;
  #pragma unroll 4
  for (int d = 0; d < D_; ++d) {
    float qv = (float)row[d];
    float kv = (float)row[C_ + d];
    float vv = (float)row[2 * C_ + d];
    sq += qv * qv;
    sk += kv * kv;
    vT[((size_t)bh * D_ + d) * T_ + t] = (__bf16)vv;
  }
  qn[idx] = sq;
  kn[idx] = sk;
}

// ---------------------------------------------------------------------------
// Generic bf16 WMMA GEMM: C[M x N] = A[M x K] * B[N x K]^T
// 128 threads = 4 waves; each wave computes a 16x64 slab (4 C fragments).
// ---------------------------------------------------------------------------
template <bool OUT_F32>
__global__ __launch_bounds__(128) void gemm_nt(const __bf16* __restrict__ A, int lda,
                                               const __bf16* __restrict__ Bm, int ldb,
                                               float* __restrict__ Cf,
                                               __bf16* __restrict__ Cb,
                                               int ldc, int K) {
  int lane = threadIdx.x & 31;
  int wave = threadIdx.x >> 5;
  int mb = blockIdx.x * 64 + wave * 16;
  int nb = blockIdx.y * 64;

  v8f acc[4] = {};
  const __bf16* arow = A + (size_t)mb * lda;

  for (int k0 = 0; k0 < K; k0 += 32) {
    v16bf a = load_frag_a(arow + k0, lda, lane);
    #pragma unroll
    for (int nt = 0; nt < 4; ++nt) {
      v16bf b = load_frag_b(Bm + (size_t)(nb + nt * 16) * ldb + k0, ldb, lane);
      acc[nt] = __builtin_amdgcn_wmma_f32_16x16x32_bf16(
          false, a, false, b, (short)0, acc[nt], false, false);
    }
  }

  int hi = lane >> 4, nlo = lane & 15;
  #pragma unroll
  for (int nt = 0; nt < 4; ++nt) {
    #pragma unroll
    for (int r = 0; r < 8; ++r) {
      size_t off = (size_t)(mb + r + 8 * hi) * ldc + nb + nt * 16 + nlo;
      if constexpr (OUT_F32) Cf[off] = acc[nt][r];
      else                   Cb[off] = (__bf16)acc[nt][r];
    }
  }
}

// ---------------------------------------------------------------------------
// Yat causal attention, flash-style online softmax.
// Grid: (B*H, T/64); 128 threads = 4 waves; each wave owns 16 q-rows.
// Scores: s = dot^2 / (|q|^2 + |k|^2 - 2 dot + eps), causal mask, softmax, P·V.
// Row-sum for the softmax denominator is computed with an extra WMMA against a
// constant all-ones B fragment (P·1), avoiding 32 ds_bpermute ops per k-tile.
// ---------------------------------------------------------------------------
__global__ __launch_bounds__(128) void yat_attn(const __bf16* __restrict__ qkv,  // [B*T, 3C]
                                                const __bf16* __restrict__ vT,   // [B*H, D, T]
                                                const float* __restrict__ qn,    // [B*H, T]
                                                const float* __restrict__ kn,    // [B*H, T]
                                                __bf16* __restrict__ attO) {     // [B*T, C]
  __shared__ __align__(16) __bf16 lds_p[4][16][32];

  int lane = threadIdx.x & 31;
  int wave = threadIdx.x >> 5;
  int bh = blockIdx.x;
  int qt = blockIdx.y;
  int b = bh / H_, h = bh % H_;
  int qr0 = qt * 64 + wave * 16;
  int hi = lane >> 4, nlo = lane & 15;

  // q A-fragments for this wave's 16 rows (d 0..31 and 32..63), lda = 3C
  const __bf16* qbase = qkv + (size_t)(b * T_ + qr0) * (3 * C_) + h * D_;
  v16bf aq0 = load_frag_a(qbase, 3 * C_, lane);
  v16bf aq1 = load_frag_a(qbase + 32, 3 * C_, lane);

  // constant all-ones B fragment: wmma(P, ones, 0) -> every column = rowsum(P)
  v16bf ones;
  #pragma unroll
  for (int i = 0; i < 16; ++i) ones[i] = (__bf16)1.0f;

  float qn_r[8];
  #pragma unroll
  for (int r = 0; r < 8; ++r)
    qn_r[r] = qn[(size_t)bh * T_ + qr0 + r + 8 * hi];

  float m[8], l[8];
  v8f acc[4] = {};
  #pragma unroll
  for (int r = 0; r < 8; ++r) { m[r] = -1e30f; l[r] = 0.f; }

  const int qlast = qr0 + 15;
  const __bf16* kcol  = qkv + (size_t)(b * T_) * (3 * C_) + C_ + h * D_;
  const __bf16* vbase = vT + (size_t)bh * D_ * T_;
  const float*  knrow = kn + (size_t)bh * T_;

  for (int j0 = 0; j0 <= qlast; j0 += 32) {
    if (j0 + 32 <= qlast)  // speculative prefetch of next k tile
      __builtin_prefetch(kcol + (size_t)(j0 + 32) * (3 * C_), 0, 1);

    // two 16-wide score tiles (cols j0.. and j0+16..), each dot over K=64
    v8f s0 = {}, s1 = {};
    {
      const __bf16* kb0 = kcol + (size_t)j0 * (3 * C_);
      v16bf b00 = load_frag_b(kb0,      3 * C_, lane);
      v16bf b01 = load_frag_b(kb0 + 32, 3 * C_, lane);
      s0 = __builtin_amdgcn_wmma_f32_16x16x32_bf16(false, aq0, false, b00, (short)0, s0, false, false);
      s0 = __builtin_amdgcn_wmma_f32_16x16x32_bf16(false, aq1, false, b01, (short)0, s0, false, false);
      const __bf16* kb1 = kcol + (size_t)(j0 + 16) * (3 * C_);
      v16bf b10 = load_frag_b(kb1,      3 * C_, lane);
      v16bf b11 = load_frag_b(kb1 + 32, 3 * C_, lane);
      s1 = __builtin_amdgcn_wmma_f32_16x16x32_bf16(false, aq0, false, b10, (short)0, s1, false, false);
      s1 = __builtin_amdgcn_wmma_f32_16x16x32_bf16(false, aq1, false, b11, (short)0, s1, false, false);
    }

    float kn0 = knrow[j0 + nlo];
    float kn1 = knrow[j0 + 16 + nlo];

    float alpha[8];
    #pragma unroll
    for (int r = 0; r < 8; ++r) {
      int row = qr0 + r + 8 * hi;
      int c0 = j0 + nlo, c1 = c0 + 16;
      float d0 = s0[r], d1 = s1[r];
      float sc0 = (c0 <= row) ? (d0 * d0) / (qn_r[r] + kn0 - 2.f * d0 + EPS_) : -1e30f;
      float sc1 = (c1 <= row) ? (d1 * d1) / (qn_r[r] + kn1 - 2.f * d1 + EPS_) : -1e30f;

      // row max across the 16 lanes of this half-wave
      float rm = fmaxf(sc0, sc1);
      rm = fmaxf(rm, __shfl_xor(rm, 1));
      rm = fmaxf(rm, __shfl_xor(rm, 2));
      rm = fmaxf(rm, __shfl_xor(rm, 4));
      rm = fmaxf(rm, __shfl_xor(rm, 8));

      float mnew = fmaxf(m[r], rm);
      alpha[r]   = __expf(m[r] - mnew);
      m[r]       = mnew;

      // masked entries produce exp(-1e30 - mnew) == 0 exactly
      float e0 = __expf(sc0 - mnew);
      float e1 = __expf(sc1 - mnew);

      // stage P into LDS (C-layout -> row-major), bf16
      lds_p[wave][r + 8 * hi][nlo]      = (__bf16)e0;
      lds_p[wave][r + 8 * hi][16 + nlo] = (__bf16)e1;

      #pragma unroll
      for (int c = 0; c < 4; ++c) acc[c][r] *= alpha[r];
    }
    asm volatile("s_wait_dscnt 0" ::: "memory");

    // reload P as an A-fragment (16x32)
    v16bf ap;
    {
      const __bf16* p = &lds_p[wave][nlo][hi * 8];
      BF16x16 t;
      t.h[0] = *(const v8bf*)(p);
      t.h[1] = *(const v8bf*)(p + 16);
      ap = t.v;
    }

    // row-sum via matrix op: rs = P · ones; every column holds rowsum(P)
    v8f rsz = {};
    v8f rs = __builtin_amdgcn_wmma_f32_16x16x32_bf16(
        false, ap, false, ones, (short)0, rsz, false, false);
    #pragma unroll
    for (int r = 0; r < 8; ++r) l[r] = l[r] * alpha[r] + rs[r];

    // acc[c] += P(16x32) * V(32x16) for each 16-wide d chunk
    #pragma unroll
    for (int c = 0; c < 4; ++c) {
      v16bf vb = load_frag_b(vbase + (size_t)(c * 16) * T_ + j0, T_, lane);
      acc[c] = __builtin_amdgcn_wmma_f32_16x16x32_bf16(
          false, ap, false, vb, (short)0, acc[c], false, false);
    }
  }

  // normalize rows and store bf16 for the output projection GEMM
  #pragma unroll
  for (int r = 0; r < 8; ++r) {
    float inv = 1.f / l[r];
    int row = qr0 + r + 8 * hi;
    #pragma unroll
    for (int c = 0; c < 4; ++c)
      attO[(size_t)(b * T_ + row) * C_ + h * D_ + c * 16 + nlo] =
          (__bf16)(acc[c][r] * inv);
  }
}

// ---------------------------------------------------------------------------
// Host launcher
// ---------------------------------------------------------------------------
extern "C" void kernel_launch(void* const* d_in, const int* in_sizes, int n_in,
                              void* d_out, int out_size, void* d_ws, size_t ws_size,
                              hipStream_t stream) {
  const float* x    = (const float*)d_in[0];  // [B,T,C]
  const float* Wqkv = (const float*)d_in[1];  // [C,3C]
  const float* Wout = (const float*)d_in[2];  // [C,C]
  float* out = (float*)d_out;                 // [B,T,C] fp32

  const int M  = B_ * T_;        // 4096
  const int K  = C_;             // 1024
  const int N1 = 3 * C_;         // 3072

  char* ws = (char*)d_ws;
  size_t off = 0;
  auto alloc = [&](size_t bytes) -> void* {
    void* p = ws + off;
    off += (bytes + 255) & ~(size_t)255;
    return p;
  };

  __bf16* xbf   = (__bf16*)alloc((size_t)M * K * 2);          // 8 MB
  __bf16* WqkvT = (__bf16*)alloc((size_t)N1 * K * 2);         // 6 MB
  __bf16* WoutT = (__bf16*)alloc((size_t)C_ * C_ * 2);        // 2 MB
  __bf16* qkvb  = (__bf16*)alloc((size_t)M * N1 * 2);         // 24 MB
  __bf16* vT    = (__bf16*)alloc((size_t)B_ * H_ * D_ * T_ * 2); // 8 MB
  float*  qn    = (float*)alloc((size_t)B_ * H_ * T_ * 4);
  float*  kn    = (float*)alloc((size_t)B_ * H_ * T_ * 4);
  __bf16* attO  = (__bf16*)alloc((size_t)M * C_ * 2);         // 8 MB

  // 1) convert / transpose operands to bf16
  int nx = M * K;
  f32_to_bf16<<<(nx + 255) / 256, 256, 0, stream>>>(x, xbf, nx);
  transpose_to_bf16<<<(K * N1 + 255) / 256, 256, 0, stream>>>(Wqkv, WqkvT, K, N1);
  transpose_to_bf16<<<(K * C_ + 255) / 256, 256, 0, stream>>>(Wout, WoutT, K, C_);

  // 2) qkv = x @ Wqkv   (A[M,K] * B[N,K]^T), bf16 out
  gemm_nt<false><<<dim3(M / 64, N1 / 64), 128, 0, stream>>>(
      xbf, K, WqkvT, K, nullptr, qkvb, N1, K);

  // 3) per-head norms + v transpose
  prep_heads<<<(B_ * H_ * T_ + 255) / 256, 256, 0, stream>>>(qkvb, vT, qn, kn);

  // 4) Yat causal attention
  yat_attn<<<dim3(B_ * H_, T_ / 64), 128, 0, stream>>>(qkvb, vT, qn, kn, attO);

  // 5) out = attO @ Wout, fp32 out
  gemm_nt<true><<<dim3(M / 64, C_ / 64), 128, 0, stream>>>(
      attO, C_, WoutT, K, out, nullptr, C_, K);
}
